// Net_31044023616357
// MI455X (gfx1250) — matrix operand order repaired and checked by reading, not compiled, not run on hardware
//
#include <hip/hip_runtime.h>
#include <math.h>

// ---------------------------------------------------------------------------
// CDNA5 WMMA types (wave32): A/B = 16 bf16 per lane (8 VGPRs), C/D = 8 f32.
// ---------------------------------------------------------------------------
typedef __attribute__((ext_vector_type(16))) __bf16 v16bf;
typedef __attribute__((ext_vector_type(8)))  float  v8f;

union AFrag { v16bf v; unsigned short h[16]; };
union BFrag { v16bf v; uint4 q[2]; unsigned short h[16]; };

__device__ inline unsigned short f2bf(float f) {
    unsigned u = __builtin_bit_cast(unsigned, f);
    unsigned r = (u + 0x7FFFu + ((u >> 16) & 1u)) >> 16;   // RNE
    return (unsigned short)r;
}
__device__ inline int imin(int a, int b) { return a < b ? a : b; }
__device__ inline int imax(int a, int b) { return a > b ? a : b; }

__device__ inline float act_apply(float v, int act) {
    if (act == 1) return fmaxf(v, 0.f);            // relu
    if (act == 2) return v > 0.f ? v : 0.1f * v;   // lrelu(0.1)
    if (act == 3) return tanhf(v);                 // tanh
    return v;
}

// Edge-clamped bilinear (matches reference bilinear_hw).
__device__ inline float bilin(const float* __restrict__ p, int H, int W, float y, float x) {
    y = fminf(fmaxf(y, 0.f), (float)(H - 1));
    x = fminf(fmaxf(x, 0.f), (float)(W - 1));
    float yf = floorf(y), xf = floorf(x);
    int y0 = (int)yf, x0 = (int)xf;
    int y1 = imin(y0 + 1, H - 1), x1 = imin(x0 + 1, W - 1);
    float wy = y - yf, wx = x - xf;
    float v00 = p[(size_t)y0 * W + x0], v01 = p[(size_t)y0 * W + x1];
    float v10 = p[(size_t)y1 * W + x0], v11 = p[(size_t)y1 * W + x1];
    return v00 * (1.f - wy) * (1.f - wx) + v01 * (1.f - wy) * wx
         + v10 * wy * (1.f - wx)         + v11 * wy * wx;
}

// ---------------------------------------------------------------------------
// Weight pack: fp32 [Cout][K] (natural trailing flatten) -> bf16 [Cout][Kpad]
// ---------------------------------------------------------------------------
__global__ void pack_weights_k(const float* __restrict__ w, unsigned short* __restrict__ wp,
                               int Cout, int K, int Kpad) {
    int idx = blockIdx.x * 256 + threadIdx.x;
    if (idx >= Cout * Kpad) return;
    int o = idx / Kpad, k = idx - o * Kpad;
    float v = (k < K) ? w[(size_t)o * K + k] : 0.f;
    wp[idx] = f2bf(v);
}

// ---------------------------------------------------------------------------
// Implicit-GEMM conv. Kernel shape, pad mode and N-tile count are compile-time:
// K-decode divisions lower to magic multiplies, and the NT accumulators are
// unconditional SSA values (no runtime guards -> no register shuffles).
// B (weight) tile staged through LDS, double-buffered, shared by all 8 waves.
// in  : fp32 [Cin][T][Hi][Wi]   wp : bf16 [Cout][Kpad]   out: [Cout][T][Ho][Wo]
// ---------------------------------------------------------------------------
template<int KD, int KH, int KW, int PADMODE, int NT>
__global__ __launch_bounds__(256) void conv_wmma_k(
    const float* __restrict__ in, const unsigned short* __restrict__ wp,
    const float* __restrict__ bias, float* __restrict__ out,
    int Cin, int T, int Hi, int Wi, int Cout, int Ho, int Wo,
    int sh, int sw, int pd, int ph, int pw, int act, int Kreal, int Kpad)
{
    constexpr int KHW  = KH * KW;
    constexpr int KDHW = KD * KHW;
    __shared__ uint4 lb[2][64 * 4];            // 2 x (64 rows x 32 bf16) = 8 KB

    int lane = threadIdx.x & 31;
    int wv   = threadIdx.x >> 5;
    int t    = blockIdx.y;
    int M    = Ho * Wo;
    int Mbase  = blockIdx.x * 128 + wv * 16;
    int CoBase = blockIdx.z * 64;

    int m = Mbase + (lane & 15);
    bool mvalid = m < M;
    int oy = 0, ox = 0;
    if (mvalid) { oy = m / Wo; ox = m - oy * Wo; }
    int kbA = (lane & 16) ? 16 : 0;
    int bsel = (lane & 16) ? 2 : 0;            // uint4 offset for K 16..31 half
    int nlo = lane & 15;
    size_t planeT = (size_t)Hi * Wi;

    int srow = threadIdx.x >> 2;               // staging: 4 threads per row
    int scol = threadIdx.x & 3;

    v8f zero = {0.f,0.f,0.f,0.f,0.f,0.f,0.f,0.f};
    v8f acc[NT];
    #pragma unroll
    for (int q = 0; q < NT; q++) acc[q] = zero;

    for (int k0 = 0; k0 < Kpad; k0 += 32) {
        int ibuf = (k0 >> 5) & 1;
        // ---- cooperative B-tile stage: 64x32 bf16 -> LDS (zero-padded rows) --
        uint4 sv = make_uint4(0, 0, 0, 0);
        if (CoBase + srow < Cout)
            sv = *reinterpret_cast<const uint4*>(wp + (size_t)(CoBase + srow) * Kpad + k0 + scol * 8);
        lb[ibuf][srow * 4 + scol] = sv;

        // ---- A fragment: per-lane im2col gather (fp32 -> bf16) ----
        AFrag a;
        #pragma unroll
        for (int j = 0; j < 16; j++) {
            int kk = k0 + kbA + j;
            float v = 0.f;
            if (mvalid && kk < Kreal) {
                int ci = kk / KDHW;
                int r  = kk - ci * KDHW;
                int dz = r / KHW; r -= dz * KHW;
                int dy = r / KW;  int dx = r - dy * KW;
                int tz = t - pd + dz;
                int iy = oy * sh - ph + dy;
                int ix = ox * sw - pw + dx;
                if (PADMODE) {                   // replicate (edge) pad
                    tz = imin(imax(tz, 0), T - 1);
                    iy = imin(imax(iy, 0), Hi - 1);
                    ix = imin(imax(ix, 0), Wi - 1);
                    v = in[((size_t)ci * T + tz) * planeT + (size_t)iy * Wi + ix];
                } else if (tz >= 0 && tz < T && iy >= 0 && iy < Hi && ix >= 0 && ix < Wi) {
                    v = in[((size_t)ci * T + tz) * planeT + (size_t)iy * Wi + ix];
                }
            }
            a.h[j] = f2bf(v);
        }

        __syncthreads();                        // B tile visible to all waves
        #pragma unroll
        for (int q = 0; q < NT; q++) {
            const uint4* lrow = &lb[ibuf][(q * 16 + nlo) * 4 + bsel];
            BFrag b; b.q[0] = lrow[0]; b.q[1] = lrow[1];
            acc[q] = __builtin_amdgcn_wmma_f32_16x16x32_bf16(
                        false, a.v, false, b.v, (short)0, acc[q], false, false);
        }
    }

    int rowOff = (lane & 16) ? 8 : 0;
    #pragma unroll
    for (int q = 0; q < NT; q++) {
        int n = CoBase + q * 16 + nlo;
        if (n >= Cout) continue;
        float bv = bias[n];
        #pragma unroll
        for (int r = 0; r < 8; r++) {
            int mm = Mbase + rowOff + r;
            if (mm < M)
                out[((size_t)n * T + t) * M + mm] = act_apply(acc[q][r] + bv, act);
        }
    }
}

// Host-side NT dispatcher (NT values used by this net: 1, 2, 4).
template<int KD, int KH, int KW, int PM>
static void launch_conv(dim3 g, hipStream_t s,
                        const float* in, const unsigned short* wp, const float* bias, float* out,
                        int Cin, int Tt, int Hi, int Wi, int Cout, int Ho, int Wo,
                        int sh, int sw, int pd, int ph, int pw, int act, int K, int Kpad) {
    int NT = (Cout + 15) / 16; if (NT > 4) NT = 4;
    if (NT == 1)
        conv_wmma_k<KD, KH, KW, PM, 1><<<g, 256, 0, s>>>(in, wp, bias, out,
            Cin, Tt, Hi, Wi, Cout, Ho, Wo, sh, sw, pd, ph, pw, act, K, Kpad);
    else if (NT == 2)
        conv_wmma_k<KD, KH, KW, PM, 2><<<g, 256, 0, s>>>(in, wp, bias, out,
            Cin, Tt, Hi, Wi, Cout, Ho, Wo, sh, sw, pd, ph, pw, act, K, Kpad);
    else
        conv_wmma_k<KD, KH, KW, PM, 4><<<g, 256, 0, s>>>(in, wp, bias, out,
            Cin, Tt, Hi, Wi, Cout, Ho, Wo, sh, sw, pd, ph, pw, act, K, Kpad);
}

// ---------------------------------------------------------------------------
// Fused deformable 3x3x3 conv (HW offsets). GEMM: M=pixels, N=Cout=64, K=64*27.
// A elements are edge-clamped bilinear samples; B staged via LDS. 4 N-tiles,
// unconditional (Cout is always 64 here).
// x : [64][T][H][W]; off : [54][T][H][W] (tap*2+comp); wp : bf16 [64][1728]
// ---------------------------------------------------------------------------
__global__ __launch_bounds__(256) void dcn_hw_k(
    const float* __restrict__ x, const float* __restrict__ offp,
    const unsigned short* __restrict__ wp, const float* __restrict__ bias,
    float* __restrict__ out, int T, int H, int W, int Kreal, int Kpad, int act)
{
    __shared__ uint4 lb[2][64 * 4];
    int lane = threadIdx.x & 31;
    int wv   = threadIdx.x >> 5;
    int t    = blockIdx.y;
    int M    = H * W;
    int Mbase = blockIdx.x * 128 + wv * 16;

    int m = Mbase + (lane & 15);
    bool mvalid = m < M;
    int oy = 0, ox = 0;
    if (mvalid) { oy = m / W; ox = m - oy * W; }
    int kbA = (lane & 16) ? 16 : 0;
    int bsel = (lane & 16) ? 2 : 0;
    int nlo = lane & 15;
    int srow = threadIdx.x >> 2, scol = threadIdx.x & 3;

    v8f zero = {0.f,0.f,0.f,0.f,0.f,0.f,0.f,0.f};
    v8f acc[4] = {zero, zero, zero, zero};

    for (int k0 = 0; k0 < Kpad; k0 += 32) {
        int ibuf = (k0 >> 5) & 1;
        uint4 sv = *reinterpret_cast<const uint4*>(wp + (size_t)srow * Kpad + k0 + scol * 8);
        lb[ibuf][srow * 4 + scol] = sv;

        AFrag a;
        #pragma unroll
        for (int j = 0; j < 16; j++) {
            int kk = k0 + kbA + j;
            float v = 0.f;
            if (mvalid && kk < Kreal) {
                int c   = kk / 27;                     // compile-time divisors
                int tap = kk - c * 27;
                int t9  = tap / 9; int r9 = tap - t9 * 9;
                int h3  = r9 / 3;
                int dt = t9 - 1, dh = h3 - 1, dw = r9 - h3 * 3 - 1;
                int tp = imin(imax(t + dt, 0), T - 1);
                float offy = offp[((size_t)(tap * 2 + 0) * T + t) * M + m];
                float offx = offp[((size_t)(tap * 2 + 1) * T + t) * M + m];
                v = bilin(x + ((size_t)c * T + tp) * M, H, W,
                          (float)oy + (float)dh + offy,
                          (float)ox + (float)dw + offx);
            }
            a.h[j] = f2bf(v);
        }

        __syncthreads();
        #pragma unroll
        for (int q = 0; q < 4; q++) {
            const uint4* lrow = &lb[ibuf][(q * 16 + nlo) * 4 + bsel];
            BFrag b; b.q[0] = lrow[0]; b.q[1] = lrow[1];
            acc[q] = __builtin_amdgcn_wmma_f32_16x16x32_bf16(
                        false, a.v, false, b.v, (short)0, acc[q], false, false);
        }
    }

    int rowOff = (lane & 16) ? 8 : 0;
    #pragma unroll
    for (int q = 0; q < 4; q++) {
        int n = q * 16 + nlo;
        float bv = bias[n];
        #pragma unroll
        for (int r = 0; r < 8; r++) {
            int mm = Mbase + rowOff + r;
            if (mm < M)
                out[((size_t)n * T + t) * M + mm] = act_apply(acc[q][r] + bv, act);
        }
    }
}

// ---------------------------------------------------------------------------
// Fused modulated deformable 5x5 over 4 frames + 1x1 aggregation (WMMA).
// frames : [4][C][H][W]; off : [200][H][W]; mask raw : [100][H][W]
// K index = c*100 + t*25 + j ; j = (dh+2)*5 + (dw+2). Cout <= 16 (one N tile).
// ---------------------------------------------------------------------------
__global__ __launch_bounds__(256) void dcn_frames_k(
    const float* __restrict__ frames, const float* __restrict__ offp,
    const float* __restrict__ maskraw, const unsigned short* __restrict__ wp,
    const float* __restrict__ bias, float* __restrict__ out,
    int C, int H, int W, int Cout, int Kreal, int Kpad)
{
    int lane = threadIdx.x & 31;
    int wv   = threadIdx.x >> 5;
    int M    = H * W;
    int Mbase = blockIdx.x * 128 + wv * 16;

    int m = Mbase + (lane & 15);
    bool mvalid = m < M;
    int oy = 0, ox = 0;
    if (mvalid) { oy = m / W; ox = m - oy * W; }
    int kbA = (lane & 16) ? 16 : 0;
    int kbB = (lane & 16) ? 16 : 0;
    int nlo = lane & 15;

    v8f acc = {0.f,0.f,0.f,0.f,0.f,0.f,0.f,0.f};

    for (int k0 = 0; k0 < Kpad; k0 += 32) {
        AFrag a;
        #pragma unroll
        for (int j16 = 0; j16 < 16; j16++) {
            int kk = k0 + kbA + j16;
            float v = 0.f;
            if (mvalid && kk < Kreal) {
                int c  = kk / 100;                    // Tn=4, kk=25 -> constants
                int r  = kk - c * 100;
                int tf = r / 25;
                int j  = r - tf * 25;
                int h5 = j / 5;
                int dh = h5 - 2, dw = j - h5 * 5 - 2;
                int tj = tf * 25 + j;
                float offy = offp[((size_t)(tj * 2 + 0)) * M + m];
                float offx = offp[((size_t)(tj * 2 + 1)) * M + m];
                float mk = maskraw[(size_t)tj * M + m];
                mk = 1.f / (1.f + expf(-mk));
                v = bilin(frames + ((size_t)tf * C + c) * M, H, W,
                          (float)oy + (float)dh + offy,
                          (float)ox + (float)dw + offx) * mk;
            }
            a.h[j16] = f2bf(v);
        }
        BFrag b;
        if (nlo < Cout) {
            const uint4* bp = reinterpret_cast<const uint4*>(wp + (size_t)nlo * Kpad + k0 + kbB);
            b.q[0] = bp[0]; b.q[1] = bp[1];
        } else {
            b.q[0] = make_uint4(0,0,0,0); b.q[1] = make_uint4(0,0,0,0);
        }
        acc = __builtin_amdgcn_wmma_f32_16x16x32_bf16(
                false, a.v, false, b.v, (short)0, acc, false, false);
    }

    int rowOff = (lane & 16) ? 8 : 0;
    if (nlo < Cout) {
        float bv = bias[nlo];
        #pragma unroll
        for (int r = 0; r < 8; r++) {
            int mm = Mbase + rowOff + r;
            if (mm < M)
                out[(size_t)nlo * M + mm] = acc[r] + bv;
        }
    }
}

// ---------------------------------------------------------------------------
// ConvTranspose3d, kernel 3x4x4, stride (1,2,2), pad (1,1,1). Gather form.
// w : [Cin][Cout][3][4][4]
// ---------------------------------------------------------------------------
__global__ void deconv3d_k(const float* __restrict__ in, const float* __restrict__ w,
                           const float* __restrict__ bias, float* __restrict__ out,
                           int Cin, int Cout, int T, int Hi, int Wi, int Ho, int Wo, int act)
{
    int idx = blockIdx.x * 256 + threadIdx.x;
    int total = Cout * T * Ho * Wo;
    if (idx >= total) return;
    int x = idx % Wo; int r = idx / Wo;
    int y = r % Ho;   r /= Ho;
    int t = r % T;    int o = r / T;
    float s = bias[o];
    size_t planeT = (size_t)Hi * Wi;
    for (int dz = 0; dz < 3; dz++) {
        int tz = t + 1 - dz;
        if (tz < 0 || tz >= T) continue;
        for (int dy = 0; dy < 4; dy++) {
            int yy = y + 1 - dy;
            if (yy < 0 || (yy & 1)) continue;
            int iy = yy >> 1; if (iy >= Hi) continue;
            for (int dx = 0; dx < 4; dx++) {
                int xx = x + 1 - dx;
                if (xx < 0 || (xx & 1)) continue;
                int ix = xx >> 1; if (ix >= Wi) continue;
                const float* ip = in + (size_t)tz * planeT + (size_t)iy * Wi + ix;
                const float* wpp = w + ((size_t)o * 3 + dz) * 16 + dy * 4 + dx;
                for (int ci = 0; ci < Cin; ci++)
                    s += ip[(size_t)ci * T * planeT] * wpp[(size_t)ci * Cout * 48];
            }
        }
    }
    out[idx] = act_apply(s, act);
}

// ---------------------------------------------------------------------------
// Coordinate attention kernels.
// ---------------------------------------------------------------------------
__global__ void ca_pool_k(const float* __restrict__ x, float* __restrict__ y0,
                          int C, int H, int W) {
    int P = H + W;
    int idx = blockIdx.x * 256 + threadIdx.x;
    if (idx >= C * P) return;
    int c = idx / P, p = idx - c * P;
    float s = 0.f;
    if (p < H) {
        const float* row = x + ((size_t)c * H + p) * W;
        for (int w = 0; w < W; w++) s += row[w];
        s /= (float)W;
    } else {
        int w = p - H;
        const float* col = x + (size_t)c * H * W + w;
        for (int h = 0; h < H; h++) s += col[(size_t)h * W];
        s /= (float)H;
    }
    y0[idx] = s;
}

__global__ void ca_mlp1_k(const float* __restrict__ y0, const float* __restrict__ w1,
                          const float* __restrict__ b1, float* __restrict__ y1,
                          int C, int mip, int P) {
    int idx = blockIdx.x * 256 + threadIdx.x;
    if (idx >= mip * P) return;
    int mi = idx / P, p = idx - mi * P;
    float s = b1[mi];
    for (int c = 0; c < C; c++) s += w1[(size_t)mi * C + c] * y0[(size_t)c * P + p];
    y1[idx] = s * fminf(fmaxf(s + 3.f, 0.f), 6.f) * (1.f / 6.f);   // h_swish
}

__global__ void ca_mlp2_k(const float* __restrict__ y1,
                          const float* __restrict__ wh, const float* __restrict__ bh,
                          const float* __restrict__ ww, const float* __restrict__ bw,
                          float* __restrict__ ah, float* __restrict__ aw,
                          int C, int mip, int H, int W) {
    int P = H + W;
    int idx = blockIdx.x * 256 + threadIdx.x;
    if (idx >= C * P) return;
    int c = idx / P, p = idx - c * P;
    if (p < H) {
        float s = bh[c];
        for (int mi = 0; mi < mip; mi++) s += wh[(size_t)c * mip + mi] * y1[(size_t)mi * P + p];
        ah[(size_t)c * H + p] = 1.f / (1.f + expf(-s));
    } else {
        int w = p - H;
        float s = bw[c];
        for (int mi = 0; mi < mip; mi++) s += ww[(size_t)c * mip + mi] * y1[(size_t)mi * P + H + w];
        aw[(size_t)c * W + w] = 1.f / (1.f + expf(-s));
    }
}

__global__ void ca_apply_k(const float* __restrict__ x, const float* __restrict__ ah,
                           const float* __restrict__ aw, float* __restrict__ out,
                           int C, int H, int W) {
    int idx = blockIdx.x * 256 + threadIdx.x;
    if (idx >= C * H * W) return;
    int w = idx % W; int r = idx / W;
    int h = r % H;   int c = r / H;
    out[idx] = x[idx] * ah[(size_t)c * H + h] * aw[(size_t)c * W + w];
}

// ---------------------------------------------------------------------------
// Glue kernels.
// ---------------------------------------------------------------------------
__global__ void eltwise_k(const float* __restrict__ a, const float* __restrict__ b,
                          float* __restrict__ o, int n, int mode) {
    int i = blockIdx.x * 256 + threadIdx.x;
    if (i >= n) return;
    float v = a[i];
    if (mode == 3) v = v + b[i];
    else v = act_apply(v, mode);   // 0 copy, 1 relu, 2 lrelu, 3 add
    o[i] = v;
}

// out [C][T][HW] from 4 per-frame tensors [C][HW]  (vol[c][t] = frame_t[c])
__global__ void pack_vol_k(const float* f0, const float* f1, const float* f2, const float* f3,
                           float* __restrict__ out, int C, int T, int HW) {
    int idx = blockIdx.x * 256 + threadIdx.x;
    if (idx >= C * T * HW) return;
    int c = idx / (T * HW); int r = idx - c * T * HW;
    int t = r / HW;         int m = r - t * HW;
    const float* f = (t == 0) ? f0 : (t == 1) ? f1 : (t == 2) ? f2 : f3;
    out[idx] = f[(size_t)c * HW + m];
}

// out [T][C][HW] from 4 per-frame tensors [C][HW]  (frame-major pack)
__global__ void pack_frames_k(const float* f0, const float* f1, const float* f2, const float* f3,
                              float* __restrict__ out, int CHW) {
    int idx = blockIdx.x * 256 + threadIdx.x;
    if (idx >= 4 * CHW) return;
    int t = idx / CHW; int m = idx - t * CHW;
    const float* f = (t == 0) ? f0 : (t == 1) ? f1 : (t == 2) ? f2 : f3;
    out[idx] = f[m];
}

// oc[t*C+c][HW] = u[c][t][HW]   (cat(unbind(dim=2), dim=1))
__global__ void permute_ct_k(const float* __restrict__ in, float* __restrict__ out,
                             int C, int T, int HW) {
    int idx = blockIdx.x * 256 + threadIdx.x;
    if (idx >= C * T * HW) return;
    int m = idx % HW; int och = idx / HW;
    int t = och / C;  int c = och - t * C;
    out[idx] = in[((size_t)c * T + t) * HW + m];
}

// ---------------------------------------------------------------------------
// Host orchestration
// ---------------------------------------------------------------------------
extern "C" void kernel_launch(void* const* d_in, const int* in_sizes, int n_in,
                              void* d_out, int out_size, void* d_ws, size_t ws_size,
                              hipStream_t stream) {
    (void)in_sizes; (void)n_in; (void)out_size; (void)ws_size;
    const int H = 192, W = 192, HW = H * W, T = 4;
    const int H2 = 96, W2 = 96, H3 = 48, W3 = 48;

    // ---- unpack inputs in setup_inputs() flatten order ----
    int ii = 0;
    auto F = [&]() { return (const float*)d_in[ii++]; };
    const float* img[4]; for (int k = 0; k < 4; k++) img[k] = F();
    const float *c11w = F(), *c11b = F(), *c12w = F(), *c12b = F(), *c13w = F(), *c13b = F();
    struct DH { const float *ow, *ob, *w, *b; };
    DH res[5][2];
    for (int r = 0; r < 5; r++)
        for (int d = 0; d < 2; d++) { res[r][d].ow = F(); res[r][d].ob = F(); res[r][d].w = F(); res[r][d].b = F(); }
    const float *d21w = F(), *d21b = F(), *d22w = F(), *d22b = F();
    const float *c22w = F(), *c22b = F(), *c23w = F(), *c23b = F();
    struct CA { const float *w1, *b1, *wh, *bh, *ww, *bw; };
    auto getCA = [&]() { CA c; c.w1 = F(); c.b1 = F(); c.wh = F(); c.bh = F(); c.ww = F(); c.bw = F(); return c; };
    CA se1 = getCA();
    const float *prew = F(), *preb = F();
    struct DF { const float *ow, *ob, *mw, *mb, *w, *b; };
    auto getDF = [&]() { DF d; d.ow = F(); d.ob = F(); d.mw = F(); d.mb = F(); d.w = F(); d.b = F(); return d; };
    DF dcnImg = getDF(), dcnCtx = getDF();
    const float *ctx1w = F(), *ctx1b = F(), *ctx2w = F(), *ctx2b = F(), *ctx3w = F(), *ctx3b = F();
    CA ctxCA = getCA();
    const float *ctxow = F(), *ctxob = F();
    CA se = getCA();
    const float *ginw = F(), *ginb = F();
    const float *gw1[3], *gbb1[3], *gw2[3], *gbb2[3];
    for (int b = 0; b < 3; b++) { gw1[b] = F(); gbb1[b] = F(); gw2[b] = F(); gbb2[b] = F(); }
    const float *goutw = F(), *goutb = F();

    // ---- deterministic bump allocator over d_ws ----
    char* base = (char*)d_ws; size_t off = 0;
    auto alloc  = [&](size_t bytes) { void* p = base + off; off += (bytes + 255) & ~(size_t)255; return p; };
    auto allocF = [&](size_t n) { return (float*)alloc(n * sizeof(float)); };

    auto pack = [&](const float* w, int Cout, int K) {
        int Kpad = (K + 31) & ~31;
        unsigned short* wp = (unsigned short*)alloc((size_t)Cout * Kpad * 2);
        int total = Cout * Kpad;
        pack_weights_k<<<(total + 255) / 256, 256, 0, stream>>>(w, wp, Cout, K, Kpad);
        return wp;
    };

    auto conv = [&](const float* in, const float* wf, const float* bias, float* out,
                    int Cin, int Tt, int Hi, int Wi, int Cout, int Ho, int Wo,
                    int kd, int kh, int kw, int sh, int sw, int pd, int ph, int pw,
                    int padMode, int act) {
        int K = Cin * kd * kh * kw, Kpad = (K + 31) & ~31;
        const unsigned short* wp = pack(wf, Cout, K);
        dim3 g((Ho * Wo + 127) / 128, Tt, (Cout + 63) / 64);
        if (kd == 3 && kh == 7 && kw == 7 && padMode == 1)
            launch_conv<3, 7, 7, 1>(g, stream, in, wp, bias, out, Cin, Tt, Hi, Wi, Cout, Ho, Wo, sh, sw, pd, ph, pw, act, K, Kpad);
        else if (kd == 3 && kh == 3 && kw == 3 && padMode == 1)
            launch_conv<3, 3, 3, 1>(g, stream, in, wp, bias, out, Cin, Tt, Hi, Wi, Cout, Ho, Wo, sh, sw, pd, ph, pw, act, K, Kpad);
        else if (kd == 3 && kh == 3 && kw == 3 && padMode == 0)
            launch_conv<3, 3, 3, 0>(g, stream, in, wp, bias, out, Cin, Tt, Hi, Wi, Cout, Ho, Wo, sh, sw, pd, ph, pw, act, K, Kpad);
        else
            launch_conv<1, 3, 3, 0>(g, stream, in, wp, bias, out, Cin, Tt, Hi, Wi, Cout, Ho, Wo, sh, sw, pd, ph, pw, act, K, Kpad);
    };

    auto eltw = [&](const float* a, const float* b, float* o, int n, int mode) {
        eltwise_k<<<(n + 255) / 256, 256, 0, stream>>>(a, b, o, n, mode);
    };

    // ---- activation buffers ----
    float* vol   = allocF((size_t)3 * T * HW);
    float* x1    = allocF((size_t)32 * T * HW);
    float* x2    = allocF((size_t)64 * T * H2 * W2);           // later reused for u1
    float* xr    = allocF((size_t)64 * T * H3 * W3);
    float* offr  = allocF((size_t)54 * T * H3 * W3);
    float* t1    = allocF((size_t)64 * T * H3 * W3);
    float* t2    = allocF((size_t)64 * T * H3 * W3);
    float* bufA  = allocF((size_t)64 * T * HW);
    float* bufB  = allocF((size_t)64 * T * HW);
    float* bufC  = allocF((size_t)256 * HW);
    float* caY0  = allocF((size_t)256 * (H + W));
    float* caY1  = allocF((size_t)16 * (H + W));
    float* caAH  = allocF((size_t)256 * H);
    float* caAW  = allocF((size_t)256 * W);
    float* ocf   = allocF((size_t)64 * HW);
    float* off200 = allocF((size_t)200 * HW);
    float* mask100 = allocF((size_t)100 * HW);
    float* frImg = allocF((size_t)4 * 3 * HW);
    float* mid   = allocF((size_t)3 * HW);
    float* cat48 = allocF((size_t)48 * HW);
    float* ctxPk = allocF((size_t)4 * 6 * HW);
    float* cat9  = allocF((size_t)9 * HW);
    float* g16   = allocF((size_t)16 * HW);
    float* gt    = allocF((size_t)16 * HW);
    float* gt2   = allocF((size_t)16 * HW);
    float* gout  = allocF((size_t)3 * HW);

    auto coordAtt = [&](float* x, const CA& p, int C, int mip, int Hh, int Ww) {
        int P = Hh + Ww;
        ca_pool_k<<<(C * P + 255) / 256, 256, 0, stream>>>(x, caY0, C, Hh, Ww);
        ca_mlp1_k<<<(mip * P + 255) / 256, 256, 0, stream>>>(caY0, p.w1, p.b1, caY1, C, mip, P);
        ca_mlp2_k<<<(C * P + 255) / 256, 256, 0, stream>>>(caY1, p.wh, p.bh, p.ww, p.bw, caAH, caAW, C, mip, Hh, Ww);
        ca_apply_k<<<(C * Hh * Ww + 255) / 256, 256, 0, stream>>>(x, caAH, caAW, x, C, Hh, Ww);
    };
    auto dcnHW = [&](const float* x, const float* offp, const DH& p, float* out,
                     int Tt, int Hh, int Ww, int act) {
        int K = 64 * 27, Kpad = K;   // 1728, multiple of 32
        const unsigned short* wp = pack(p.w, 64, K);
        dim3 g((Hh * Ww + 127) / 128, Tt, 1);
        dcn_hw_k<<<g, 256, 0, stream>>>(x, offp, wp, p.b, out, Tt, Hh, Ww, K, Kpad, act);
    };

    // ================= forward pass =================
    pack_vol_k<<<(3 * T * HW + 255) / 256, 256, 0, stream>>>(img[0], img[1], img[2], img[3], vol, 3, T, HW);

    // encoder (replicate pad)
    conv(vol, c11w, c11b, x1, 3, T, H, W, 32, H, W,  3, 7, 7, 1, 1, 1, 3, 3, 1, 1);
    conv(x1, c12w, c12b, x2, 32, T, H, W, 64, H2, W2, 3, 3, 3, 2, 2, 1, 1, 1, 1, 1);
    conv(x2, c13w, c13b, xr, 64, T, H2, W2, 64, H3, W3, 3, 3, 3, 2, 2, 1, 1, 1, 1, 1);

    // 5 deformable residual blocks
    for (int rb = 0; rb < 5; rb++) {
        conv(xr, res[rb][0].ow, res[rb][0].ob, offr, 64, T, H3, W3, 54, H3, W3, 3, 3, 3, 1, 1, 1, 1, 1, 0, 0);
        dcnHW(xr, offr, res[rb][0], t1, T, H3, W3, 2);
        conv(t1, res[rb][1].ow, res[rb][1].ob, offr, 64, T, H3, W3, 54, H3, W3, 3, 3, 3, 1, 1, 1, 1, 1, 0, 0);
        dcnHW(t1, offr, res[rb][1], t2, T, H3, W3, 0);
        eltw(t2, xr, xr, 64 * T * H3 * W3, 3);
    }

    // decoder
    float* u1 = x2;   // reuse [64][4][96][96]
    {
        int total = 64 * T * H2 * W2;
        deconv3d_k<<<(total + 255) / 256, 256, 0, stream>>>(xr, d21w, d21b, u1, 64, 64, T, H3, W3, H2, W2, 1);
        total = 64 * T * HW;
        deconv3d_k<<<(total + 255) / 256, 256, 0, stream>>>(u1, d22w, d22b, bufA, 64, 64, T, H2, W2, H, W, 1);
    }
    conv(bufA, c22w, c22b, bufB, 64, T, H, W, 64, H, W, 3, 3, 3, 1, 1, 1, 1, 1, 1, 0);
    conv(bufB, c23w, c23b, bufA, 64, T, H, W, 64, H, W, 3, 7, 7, 1, 1, 1, 3, 3, 1, 3);

    // offset features: permute -> coord-att -> 3x3 conv
    permute_ct_k<<<(256 * HW + 255) / 256, 256, 0, stream>>>(bufA, bufC, 64, T, HW);
    coordAtt(bufC, se1, 256, 16, H, W);
    conv(bufC, prew, preb, ocf, 256, 1, H, W, 64, H, W, 1, 3, 3, 1, 1, 0, 1, 1, 0, 1);

    // deformable sampling of the input frames
    pack_frames_k<<<(4 * 3 * HW + 255) / 256, 256, 0, stream>>>(img[0], img[1], img[2], img[3], frImg, 3 * HW);
    conv(ocf, dcnImg.ow, dcnImg.ob, off200, 64, 1, H, W, 200, H, W, 1, 3, 3, 1, 1, 0, 1, 1, 0, 0);
    conv(ocf, dcnImg.mw, dcnImg.mb, mask100, 64, 1, H, W, 100, H, W, 1, 3, 3, 1, 1, 0, 1, 1, 0, 0);
    {
        int K = 3 * 100, Kpad = (K + 31) & ~31;
        const unsigned short* wp = pack(dcnImg.w, 3, K);
        dcn_frames_k<<<dim3((HW + 127) / 128), 256, 0, stream>>>(frImg, off200, mask100, wp, dcnImg.b, mid,
                                                                 3, H, W, 3, K, Kpad);
    }

    // context branches (one per frame), outputs packed [4][6][HW]
    for (int t = 0; t < 4; t++) {
        float* f1 = cat48;
        float* f2 = cat48 + (size_t)16 * HW;
        float* f3 = cat48 + (size_t)32 * HW;
        conv(img[t], ctx1w, ctx1b, f1, 3, 1, H, W, 16, H, W, 1, 3, 3, 1, 1, 0, 1, 1, 0, 2);
        conv(f1, ctx2w, ctx2b, f2, 16, 1, H, W, 16, H, W, 1, 3, 3, 1, 1, 0, 1, 1, 0, 2);
        conv(f2, ctx3w, ctx3b, f3, 16, 1, H, W, 16, H, W, 1, 3, 3, 1, 1, 0, 1, 1, 0, 2);
        coordAtt(cat48, ctxCA, 48, 16, H, W);
        conv(cat48, ctxow, ctxob, ctxPk + (size_t)t * 6 * HW, 48, 1, H, W, 6, H, W, 1, 3, 3, 1, 1, 0, 1, 1, 0, 0);
    }

    // deformable sampling of the context maps -> cat9[3..8]
    conv(ocf, dcnCtx.ow, dcnCtx.ob, off200, 64, 1, H, W, 200, H, W, 1, 3, 3, 1, 1, 0, 1, 1, 0, 0);
    conv(ocf, dcnCtx.mw, dcnCtx.mb, mask100, 64, 1, H, W, 100, H, W, 1, 3, 3, 1, 1, 0, 1, 1, 0, 0);
    {
        int K = 6 * 100, Kpad = (K + 31) & ~31;
        const unsigned short* wp = pack(dcnCtx.w, 6, K);
        dcn_frames_k<<<dim3((HW + 127) / 128), 256, 0, stream>>>(ctxPk, off200, mask100, wp, dcnCtx.b,
                                                                 cat9 + (size_t)3 * HW, 6, H, W, 6, K, Kpad);
    }
    eltw(mid, nullptr, cat9, 3 * HW, 0);   // cat9[0..2] = mid_out

    // final attention + grid net
    coordAtt(cat9, se, 9, 8, H, W);
    conv(cat9, ginw, ginb, g16, 9, 1, H, W, 16, H, W, 1, 3, 3, 1, 1, 0, 1, 1, 0, 1);
    for (int b = 0; b < 3; b++) {
        conv(g16, gw1[b], gbb1[b], gt, 16, 1, H, W, 16, H, W, 1, 3, 3, 1, 1, 0, 1, 1, 0, 1);
        conv(gt, gw2[b], gbb2[b], gt2, 16, 1, H, W, 16, H, W, 1, 3, 3, 1, 1, 0, 1, 1, 0, 0);
        eltw(gt2, g16, g16, 16 * HW, 3);
    }
    conv(g16, goutw, goutb, gout, 16, 1, H, W, 3, H, W, 1, 3, 3, 1, 1, 0, 1, 1, 0, 0);

    // output = mid_out + grid_out
    eltw(mid, gout, (float*)d_out, 3 * HW, 3);
}
